// SO2EquivariantGraphAttention_12163347382318
// MI455X (gfx1250) — matrix operand order repaired and checked by reading, not compiled, not run on hardware
//
#include <hip/hip_runtime.h>
#include <hip/hip_bf16.h>
#include <math.h>

// ---------------- problem constants ----------------
#define N_NODES_  5000
#define N_EDGES_  25000
#define NC_       25      // (L+1)^2
#define MSZ_      19      // 5 + 2*(4+3)
#define NH_       8
#define G_        100
#define RAD_OUT_  1536    // (5+4+3)*2*64

typedef __attribute__((ext_vector_type(16))) _Float16 v16h;
typedef __attribute__((ext_vector_type(8)))  float    v8f;

// ---------------- WMMA helpers (gfx1250, wave32) ----------------
__device__ __forceinline__ v8f wmma_(v16h a, v16h b, v8f c) {
  return __builtin_amdgcn_wmma_f32_16x16x32_f16(false, a, false, b, (short)0, c, false, false);
}

// A fragment: 16 (rows) x 32 (K) f16 from LDS, source pre-offset by k0. K order used here:
// lanes 0-15: row=lane, K 0..15; lanes 16-31: K 16..31 (contiguous, matches swz_idx)
__device__ __forceinline__ v16h frag_a16(const _Float16* A, int lda, int lane) {
  int row = lane & 15;
  int kb  = (lane >> 4) << 4;
  const _Float16* p = A + row * lda + kb;
  v16h r;
#pragma unroll
  for (int i = 0; i < 8; ++i) { r[i] = p[i]; r[i + 8] = p[8 + i]; }
  return r;
}

// Swizzled-fragment load: tile-major, 32 lanes x 16 halves (32B) contiguous per lane.
// One fragment = 2x b128 loads.
__device__ __forceinline__ v16h frag_swz(const _Float16* Bz, int tile, int lane) {
  return *((const v16h*)Bz + tile * 32 + lane);
}

// swizzled destination index for element (k, n) of a K x N B-matrix
// (tile = (k/32)*(N/16) + n/16 ; lane = 16*((k%32)>=16) + n%16 ; elem = k%16)
__device__ __forceinline__ int swz_idx(int k, int n, int NT) {
  int tile = (k >> 5) * NT + (n >> 4);
  int lane = (((k & 31) >> 4) << 4) + (n & 15);
  return (tile * 32 + lane) * 16 + (k & 15);
}

// Shared-weight GEMM building block: A (16 x KDIM f16, LDS) times 4 adjacent
// 16-wide column tiles of swizzled f16 weights. One A read feeds 4 WMMAs.
template <int KDIM, int NT>
__device__ __forceinline__ void gemm_a1b4(const _Float16* sA, int lda,
                                          const _Float16* Wz, int nt0,
                                          int lane, v8f acc[4]) {
#pragma unroll
  for (int k0 = 0; k0 < KDIM; k0 += 32) {
    v16h a = frag_a16(sA + k0, lda, lane);
    int kt = k0 >> 5;
#pragma unroll
    for (int j = 0; j < 4; ++j) {
      v16h b = frag_swz(Wz, kt * NT + nt0 + j, lane);
      acc[j] = wmma_(a, b, acc[j]);
    }
  }
}

__device__ __forceinline__ float silu_(float x) { return x / (1.f + __expf(-x)); }

__device__ __forceinline__ unsigned enc_f(float f) {
  unsigned u = __float_as_uint(f);
  return (u & 0x80000000u) ? ~u : (u | 0x80000000u);
}
__device__ __forceinline__ float dec_f(unsigned u) {
  unsigned v = (u & 0x80000000u) ? (u & 0x7FFFFFFFu) : ~u;
  return __uint_as_float(v);
}

// =====================================================================
// Kernel 0: weight pre-swizzle  f32 [K][N] -> f16 fragment-contiguous
// =====================================================================
__global__ void k_swz(const float* W, _Float16* dst, int K, int N) {
  int total = K * N;
  int NT = N >> 4;
  for (int idx = blockIdx.x * blockDim.x + threadIdx.x; idx < total;
       idx += gridDim.x * blockDim.x) {
    int k = idx / N, n = idx % N;
    dst[swz_idx(k, n, NT)] = (_Float16)W[idx];
  }
}

// LN over 64 feats + SiLU for a 16x64 f32 LDS tile -> f16 LDS tile (stride 64)
__device__ __forceinline__ void ln_silu_16x64(const float* H, const float* sc, const float* bi,
                                              _Float16* out, float* mu, float* rs, int lane) {
  if (lane < 16) {
    float m = 0.f;
#pragma unroll
    for (int c = 0; c < 64; ++c) m += H[lane * 64 + c];
    m *= (1.f / 64.f);
    float v = 0.f;
#pragma unroll
    for (int c = 0; c < 64; ++c) { float d = H[lane * 64 + c] - m; v += d * d; }
    mu[lane] = m;
    rs[lane] = rsqrtf(v * (1.f / 64.f) + 1e-6f);
  }
  __syncthreads();
  for (int idx = lane; idx < 16 * 64; idx += 32) {
    int r = idx >> 6, c = idx & 63;
    float x = (H[idx] - mu[r]) * rs[r] * sc[c] + bi[c];
    out[idx] = (_Float16)silu_(x);
  }
}

// =====================================================================
// Kernel 1: radial MLP  (192 -> 64 -> 64 -> 1536), 16 edges per wave
// =====================================================================
__global__ __launch_bounds__(32) void k_radial(
    const float* edist, const float* emb_s, const float* emb_r,
    const _Float16* w1, const float* b1, const float* ln1s, const float* ln1b,
    const _Float16* w2, const float* b2, const float* ln2s, const float* ln2b,
    const _Float16* w3, const float* b3,
    const int* species, const int* senders, const int* receivers, float* rad) {
  __shared__ _Float16 sA[16 * 192];
  __shared__ float    sH[16 * 64];
  __shared__ float    sMu[16], sRs[16];
  int lane  = threadIdx.x;
  int ebase = blockIdx.x * 16;

  for (int idx = lane; idx < 16 * 192; idx += 32) {
    int row = idx / 192, c = idx % 192;
    int e = ebase + row; if (e > N_EDGES_ - 1) e = N_EDGES_ - 1;
    float v;
    if (c < 64)       v = edist[(size_t)e * 64 + c];
    else if (c < 128) v = emb_s[(size_t)species[senders[e]]   * 64 + (c - 64)];
    else              v = emb_r[(size_t)species[receivers[e]] * 64 + (c - 128)];
    sA[idx] = (_Float16)v;
  }
  __syncthreads();

  int col16 = lane & 15, rb = (lane >> 4) << 3;

  // layer 1: 192 -> 64
  {
    v8f acc[4] = {};
    gemm_a1b4<192, 4>(sA, 192, w1, 0, lane, acc);
#pragma unroll
    for (int j = 0; j < 4; ++j) {
      int col = j * 16 + col16;
      float bias = b1[col];
#pragma unroll
      for (int g = 0; g < 8; ++g) sH[(rb + g) * 64 + col] = acc[j][g] + bias;
    }
  }
  __syncthreads();
  ln_silu_16x64(sH, ln1s, ln1b, sA, sMu, sRs, lane);
  __syncthreads();

  // layer 2: 64 -> 64
  {
    v8f acc[4] = {};
    gemm_a1b4<64, 4>(sA, 64, w2, 0, lane, acc);
#pragma unroll
    for (int j = 0; j < 4; ++j) {
      int col = j * 16 + col16;
      float bias = b2[col];
#pragma unroll
      for (int g = 0; g < 8; ++g) sH[(rb + g) * 64 + col] = acc[j][g] + bias;
    }
  }
  __syncthreads();
  ln_silu_16x64(sH, ln2s, ln2b, sA, sMu, sRs, lane);
  __syncthreads();

  // layer 3: 64 -> 1536 (to global)
  for (int ng = 0; ng < 24; ++ng) {
    v8f acc[4] = {};
    gemm_a1b4<64, 96>(sA, 64, w3, ng * 4, lane, acc);
#pragma unroll
    for (int j = 0; j < 4; ++j) {
      int col = ng * 64 + j * 16 + col16;
      float bias = b3[col];
#pragma unroll
      for (int g = 0; g < 8; ++g) {
        int e = ebase + rb + g;
        if (e < N_EDGES_) rad[(size_t)e * RAD_OUT_ + col] = acc[j][g] + bias;
      }
    }
  }
}

// =====================================================================
// Kernel 2: edge_feats = wigner[e] (19x25) @ concat(nf[s], nf[r]) (25x128)
// one wave per edge; B staged in swizzled fragment layout
// =====================================================================
__global__ __launch_bounds__(32) void k_wigner(
    const float* nf, const float* wig, const int* senders, const int* receivers, float* ef1) {
  __shared__ alignas(32) _Float16 sA[32 * 32];
  __shared__ alignas(32) _Float16 sB[8 * 32 * 16];   // 8 tiles of 32x16
  int lane = threadIdx.x;
  int e = blockIdx.x;
  int s = senders[e], r = receivers[e];
  for (int i = lane; i < 32 * 32;      i += 32) sA[i] = (_Float16)0.f;
  for (int i = lane; i < 8 * 32 * 16;  i += 32) sB[i] = (_Float16)0.f;
  __syncthreads();
  for (int i = lane; i < MSZ_ * NC_; i += 32) {
    int m = i / NC_, k = i % NC_;
    sA[m * 32 + k] = (_Float16)wig[(size_t)e * (MSZ_ * NC_) + i];
  }
  for (int i = lane; i < NC_ * 128; i += 32) {
    int k = i >> 7, c = i & 127;
    float v = (c < 64) ? nf[((size_t)s * NC_ + k) * 64 + c]
                       : nf[((size_t)r * NC_ + k) * 64 + (c - 64)];
    sB[swz_idx(k, c, 8)] = (_Float16)v;
  }
  __syncthreads();
#pragma unroll
  for (int mt = 0; mt < 2; ++mt) {
    v16h a = frag_a16(sA + mt * 16 * 32, 32, lane);
#pragma unroll
    for (int nt = 0; nt < 8; ++nt) {
      v8f acc = {};
      v16h b = frag_swz(sB, nt, lane);
      acc = wmma_(a, b, acc);
      int col = nt * 16 + (lane & 15);
      int rbm = mt * 16 + ((lane >> 4) << 3);
#pragma unroll
      for (int g = 0; g < 8; ++g) {
        int m = rbm + g;
        if (m < MSZ_) ef1[((size_t)e * MSZ_ + m) * 128 + col] = acc[g];
      }
    }
  }
}

// =====================================================================
// SO2 m>0 path (fully templated, swizzled weights):
// out0 = x0@W[:, :H] - x1@W[:, H:], out1 = x1@W[:, :H] + x0@W[:, H:]
// =====================================================================
template <int CIN, int KDIM, int H, int COUT, int ROWOFF, int NROT, bool HASRAD>
__device__ __forceinline__ void so2_m_path(const float* ef_in, const float* rad, int radoff,
                                           const _Float16* Wz, float* ef_out,
                                           int ebase, int lane, _Float16* sX) {
  __syncthreads();
  for (int idx = lane; idx < 16 * KDIM; idx += 32) {
    int row = idx / KDIM, i = idx % KDIM;
    int e = ebase + row; if (e > N_EDGES_ - 1) e = N_EDGES_ - 1;
    int m0 = ROWOFF + i / CIN, c = i % CIN;
    float rv = HASRAD ? rad[(size_t)e * RAD_OUT_ + radoff + i] : 1.f;
    sX[row * KDIM + i] =
        (_Float16)(ef_in[((size_t)e * MSZ_ + m0) * CIN + c] * rv);
    sX[16 * KDIM + row * KDIM + i] =
        (_Float16)(ef_in[((size_t)e * MSZ_ + m0 + NROT) * CIN + c] * rv);
  }
  __syncthreads();
  const int NT = (2 * H) >> 4;
#pragma unroll 1
  for (int nt = 0; nt < H / 16; ++nt) {
    v8f aRL = {}, aRR = {}, aIL = {}, aIR = {};
#pragma unroll
    for (int k0 = 0; k0 < KDIM; k0 += 32) {
      int kt = k0 >> 5;
      v16h a0 = frag_a16(sX + k0, KDIM, lane);
      v16h a1 = frag_a16(sX + 16 * KDIM + k0, KDIM, lane);
      v16h bl = frag_swz(Wz, kt * NT + nt, lane);
      v16h br = frag_swz(Wz, kt * NT + (H >> 4) + nt, lane);
      aRL = wmma_(a0, bl, aRL);
      aIR = wmma_(a1, br, aIR);
      aIL = wmma_(a1, bl, aIL);
      aRR = wmma_(a0, br, aRR);
    }
    v8f o0 = aRL - aIR;
    v8f o1 = aIL + aRR;
    int j  = nt * 16 + (lane & 15);
    int rb = (lane >> 4) << 3;
    int mo = j / COUT, c = j % COUT;
#pragma unroll
    for (int g = 0; g < 8; ++g) {
      int e = ebase + rb + g;
      if (e >= N_EDGES_) continue;
      ef_out[((size_t)e * MSZ_ + ROWOFF + mo)        * COUT + c] = o0[g];
      ef_out[((size_t)e * MSZ_ + ROWOFF + NROT + mo) * COUT + c] = o1[g];
    }
  }
}

// =====================================================================
// Kernel 3: SO2 conv #1 (c_in=128, c_out=64, rad-scaled, extra=320)
// =====================================================================
__global__ __launch_bounds__(32) void k_c1(
    const float* ef1, const float* rad, const _Float16* w0, const float* b0,
    const _Float16* w1, const _Float16* w2,
    float* efm, float* x0a, float* x0g) {
  __shared__ _Float16 sX[16 * 1024];
  int lane  = threadIdx.x;
  int ebase = blockIdx.x * 16;
  for (int idx = lane; idx < 16 * 640; idx += 32) {
    int row = idx / 640, i = idx % 640;
    int e = ebase + row; if (e > N_EDGES_ - 1) e = N_EDGES_ - 1;
    float v = ef1[((size_t)e * MSZ_ + (i >> 7)) * 128 + (i & 127)] *
              rad[(size_t)e * RAD_OUT_ + i];
    sX[row * 640 + i] = (_Float16)v;
  }
  __syncthreads();
  int col16 = lane & 15, rb = (lane >> 4) << 3;
#pragma unroll 1
  for (int ng = 0; ng < 10; ++ng) {
    v8f acc[4] = {};
    gemm_a1b4<640, 40>(sX, 640, w0, ng * 4, lane, acc);
#pragma unroll
    for (int j = 0; j < 4; ++j) {
      int col = ng * 64 + j * 16 + col16;
      float bias = b0[col];
#pragma unroll
      for (int g = 0; g < 8; ++g) {
        int e = ebase + rb + g;
        if (e >= N_EDGES_) continue;
        float v = acc[j][g] + bias;
        if (col < 320)      efm[((size_t)e * MSZ_ + (col >> 6)) * 64 + (col & 63)] = v;
        else if (col < 576) x0a[(size_t)e * 256 + (col - 320)] = v;
        else                x0g[(size_t)e * 64 + (col - 576)] = v;
      }
    }
  }
  so2_m_path<128, 512, 256, 64,  5, 4, true>(ef1, rad,  640, w1, efm, ebase, lane, sX);
  so2_m_path<128, 384, 192, 64, 13, 3, true>(ef1, rad, 1152, w2, efm, ebase, lane, sX);
}

// =====================================================================
// Kernel 4: grid round-trip + gating row, one wave per edge, in-place on efm
// sE and sG held in swizzled fragment layout
// =====================================================================
__global__ __launch_bounds__(32) void k_grid(
    const float* tg, const float* fg, const float* x0g, float* efm) {
  __shared__ alignas(32) _Float16 sE[4 * 32 * 16];     // B of GEMM1: (19->32) x 64
  __shared__ alignas(32) _Float16 sG[16 * 32 * 16];    // B of GEMM2: (100->128) x 64
  __shared__ alignas(32) _Float16 sTG[112 * 32];       // to_grid (100->112) x (19->32)
  __shared__ alignas(32) _Float16 sFG[32 * 128];       // from_grid^T (19->32) x (100->128)
  int lane = threadIdx.x;
  int e = blockIdx.x;
  for (int i = lane; i < 4 * 32 * 16;  i += 32) sE[i]  = (_Float16)0.f;
  for (int i = lane; i < 16 * 32 * 16; i += 32) sG[i]  = (_Float16)0.f;
  for (int i = lane; i < 112 * 32;     i += 32) sTG[i] = (_Float16)0.f;
  for (int i = lane; i < 32 * 128;     i += 32) sFG[i] = (_Float16)0.f;
  __syncthreads();
  for (int i = lane; i < MSZ_ * 64; i += 32) {
    int m = i >> 6, c = i & 63;
    sE[swz_idx(m, c, 4)] = (_Float16)efm[(size_t)e * MSZ_ * 64 + i];
  }
  for (int i = lane; i < G_ * MSZ_; i += 32) {
    int g = i / MSZ_, m = i % MSZ_;
    sTG[g * 32 + m]  = (_Float16)tg[i];
    sFG[m * 128 + g] = (_Float16)fg[i];
  }
  __syncthreads();
  // grid = silu(TG @ E) -> stored swizzled as B of GEMM2
#pragma unroll 1
  for (int mt = 0; mt < 7; ++mt) {
    v16h a = frag_a16(sTG + mt * 16 * 32, 32, lane);
#pragma unroll
    for (int nt = 0; nt < 4; ++nt) {
      v8f acc = {};
      v16h b = frag_swz(sE, nt, lane);
      acc = wmma_(a, b, acc);
      int nn  = lane & 15;
      int rbm = mt * 16 + ((lane >> 4) << 3);
#pragma unroll
      for (int g = 0; g < 8; ++g) {
        int k = rbm + g;  // grid row = K of GEMM2
        sG[swz_idx(k, nt * 16 + nn, 4)] = (_Float16)silu_(acc[g]);
      }
    }
  }
  __syncthreads();
  // feats = FG^T @ grid ; rows 1..18 back to efm
#pragma unroll 1
  for (int mt = 0; mt < 2; ++mt) {
#pragma unroll
    for (int nt = 0; nt < 4; ++nt) {
      v8f acc = {};
#pragma unroll
      for (int kt = 0; kt < 4; ++kt) {
        v16h a = frag_a16(sFG + mt * 16 * 128 + kt * 32, 128, lane);
        v16h b = frag_swz(sG, kt * 4 + nt, lane);
        acc = wmma_(a, b, acc);
      }
      int col = nt * 16 + (lane & 15);
      int rbm = mt * 16 + ((lane >> 4) << 3);
#pragma unroll
      for (int g = 0; g < 8; ++g) {
        int m = rbm + g;
        if (m >= 1 && m < MSZ_) efm[((size_t)e * MSZ_ + m) * 64 + col] = acc[g];
      }
    }
  }
  for (int c = lane; c < 64; c += 32)
    efm[(size_t)e * MSZ_ * 64 + c] = silu_(x0g[(size_t)e * 64 + c]);
}

// =====================================================================
// Kernel 5: SO2 conv #2 (c_in=64, c_out=128, no rad, no extra)
// =====================================================================
__global__ __launch_bounds__(32) void k_c2(
    const float* efm, const _Float16* w0, const float* b0,
    const _Float16* w1, const _Float16* w2, float* ef2) {
  __shared__ _Float16 sX[16 * 1024];
  int lane  = threadIdx.x;
  int ebase = blockIdx.x * 16;
  for (int idx = lane; idx < 16 * 320; idx += 32) {
    int row = idx / 320, i = idx % 320;
    int e = ebase + row; if (e > N_EDGES_ - 1) e = N_EDGES_ - 1;
    sX[row * 320 + i] = (_Float16)efm[((size_t)e * MSZ_ + (i >> 6)) * 64 + (i & 63)];
  }
  __syncthreads();
  int col16 = lane & 15, rb = (lane >> 4) << 3;
#pragma unroll 1
  for (int ng = 0; ng < 10; ++ng) {
    v8f acc[4] = {};
    gemm_a1b4<320, 40>(sX, 320, w0, ng * 4, lane, acc);
#pragma unroll
    for (int j = 0; j < 4; ++j) {
      int col = ng * 64 + j * 16 + col16;
      float bias = b0[col];
#pragma unroll
      for (int g = 0; g < 8; ++g) {
        int e = ebase + rb + g;
        if (e >= N_EDGES_) continue;
        ef2[((size_t)e * MSZ_ + (col >> 7)) * 128 + (col & 127)] = acc[j][g] + bias;
      }
    }
  }
  so2_m_path<64, 256, 512, 128,  5, 4, false>(efm, nullptr, 0, w1, ef2, ebase, lane, sX);
  so2_m_path<64, 192, 384, 128, 13, 3, false>(efm, nullptr, 0, w2, ef2, ebase, lane, sX);
}

// =====================================================================
// Kernel 6: init node accumulators + segment softmax stats
// =====================================================================
__global__ void k_init(float* node_out, unsigned* nmax, float* nden) {
  size_t t = (size_t)blockIdx.x * blockDim.x + threadIdx.x;
  size_t NO = (size_t)N_NODES_ * NC_ * 128;
  for (size_t i = t; i < NO; i += (size_t)gridDim.x * blockDim.x) node_out[i] = 0.f;
  if (t < (size_t)N_NODES_ * NH_) { nmax[t] = enc_f(-1e30f); nden[t] = 0.f; }
}

__global__ void k_alpha(const float* x0a, const float* an_s, const float* an_b,
                        const float* adot, const int* receivers,
                        float* a_ex, unsigned* nmax) {
  int t = blockIdx.x * blockDim.x + threadIdx.x;
  if (t >= N_EDGES_ * NH_) return;
  int e = t >> 3, h = t & 7;
  const float* x = x0a + (size_t)e * 256 + h * 32;
  float mu = 0.f;
#pragma unroll
  for (int i = 0; i < 32; ++i) mu += x[i];
  mu *= (1.f / 32.f);
  float var = 0.f;
#pragma unroll
  for (int i = 0; i < 32; ++i) { float d = x[i] - mu; var += d * d; }
  float rs = rsqrtf(var * (1.f / 32.f) + 1e-6f);
  float acc = 0.f;
#pragma unroll
  for (int i = 0; i < 32; ++i) {
    float xv = (x[i] - mu) * rs * an_s[i] + an_b[i];
    float sg = 1.f / (1.f + __expf(-xv));
    float sl = 0.6f * xv + 0.4f * xv * (2.f * sg - 1.f);
    acc += sl * adot[h * 32 + i];
  }
  a_ex[t] = acc;
  atomicMax(&nmax[receivers[e] * NH_ + h], enc_f(acc));
}

__global__ void k_expsum(const int* receivers, const unsigned* nmax,
                         float* a_ex, float* nden) {
  int t = blockIdx.x * blockDim.x + threadIdx.x;
  if (t >= N_EDGES_ * NH_) return;
  int e = t >> 3, h = t & 7;
  float amax = dec_f(nmax[receivers[e] * NH_ + h]);
  if (!(amax > -1e29f && amax < 1e29f)) amax = 0.f;
  float ex = __expf(a_ex[t] - amax);
  a_ex[t] = ex;
  atomicAdd(&nden[receivers[e] * NH_ + h], ex);
}

// =====================================================================
// Kernel 7: edge_out = wigner_inv[e] (25x19) @ (attn-weighted ef2) ; scatter-add
// =====================================================================
__global__ __launch_bounds__(32) void k_attn(
    const float* ef2, const float* winv, const float* a_ex, const float* nden,
    const int* receivers, float* node_out) {
  __shared__ alignas(32) _Float16 sW[32 * 32];
  __shared__ alignas(32) _Float16 sT[8 * 32 * 16];
  __shared__ float sAl[NH_];
  int lane = threadIdx.x;
  int e = blockIdx.x;
  int rn = receivers[e];
  if (lane < NH_) sAl[lane] = a_ex[(size_t)e * NH_ + lane] /
                              (nden[(size_t)rn * NH_ + lane] + 1e-16f);
  for (int i = lane; i < 32 * 32;     i += 32) sW[i] = (_Float16)0.f;
  for (int i = lane; i < 8 * 32 * 16; i += 32) sT[i] = (_Float16)0.f;
  __syncthreads();
  for (int i = lane; i < NC_ * MSZ_; i += 32) {
    int k = i / MSZ_, m = i % MSZ_;
    sW[k * 32 + m] = (_Float16)winv[(size_t)e * NC_ * MSZ_ + i];
  }
  for (int i = lane; i < MSZ_ * 128; i += 32) {
    int m = i >> 7, c = i & 127;
    sT[swz_idx(m, c, 8)] =
        (_Float16)(ef2[((size_t)e * MSZ_ + m) * 128 + c] * sAl[c >> 4]);
  }
  __syncthreads();
#pragma unroll
  for (int mt = 0; mt < 2; ++mt) {
    v16h a = frag_a16(sW + mt * 16 * 32, 32, lane);
#pragma unroll
    for (int nt = 0; nt < 8; ++nt) {
      v8f acc = {};
      v16h b = frag_swz(sT, nt, lane);
      acc = wmma_(a, b, acc);
      int col = nt * 16 + (lane & 15);
      int rbm = mt * 16 + ((lane >> 4) << 3);
#pragma unroll
      for (int g = 0; g < 8; ++g) {
        int k = rbm + g;
        if (k < NC_) atomicAdd(&node_out[((size_t)rn * NC_ + k) * 128 + col], acc[g]);
      }
    }
  }
}

// =====================================================================
// Kernel 8: per-m projection  out[n,m,:] = node_out[n,m,:] @ proj_w[l(m)] (+bias at m=0)
// =====================================================================
__global__ __launch_bounds__(32) void k_proj(
    const float* node_out, const _Float16* pwz, const float* pb, float* out) {
  __shared__ _Float16 sA[16 * 128];
  int lane  = threadIdx.x;
  int nbase = blockIdx.x * 16;
  int col16 = lane & 15, rb = (lane >> 4) << 3;
#pragma unroll 1
  for (int m = 0; m < NC_; ++m) {
    int l = (m == 0) ? 0 : (m < 4) ? 1 : (m < 9) ? 2 : (m < 16) ? 3 : 4;
    __syncthreads();
    for (int idx = lane; idx < 16 * 128; idx += 32) {
      int row = idx >> 7, c = idx & 127;
      int nn = nbase + row; if (nn > N_NODES_ - 1) nn = N_NODES_ - 1;
      sA[idx] = (_Float16)node_out[((size_t)nn * NC_ + m) * 128 + c];
    }
    __syncthreads();
    const _Float16* Wz = pwz + (size_t)l * 128 * 64;
    v8f acc[4] = {};
    gemm_a1b4<128, 4>(sA, 128, Wz, 0, lane, acc);
#pragma unroll
    for (int j = 0; j < 4; ++j) {
      int col = j * 16 + col16;
      float bias = (m == 0) ? pb[col] : 0.f;
#pragma unroll
      for (int g = 0; g < 8; ++g) {
        int nn = nbase + rb + g;
        if (nn < N_NODES_) out[((size_t)nn * NC_ + m) * 64 + col] = acc[j][g] + bias;
      }
    }
  }
}

// =====================================================================
extern "C" void kernel_launch(void* const* d_in, const int* in_sizes, int n_in,
                              void* d_out, int out_size, void* d_ws, size_t ws_size,
                              hipStream_t stream) {
  const float* node_feats = (const float*)d_in[0];
  const float* edist      = (const float*)d_in[1];
  const float* wig        = (const float*)d_in[2];
  const float* winv       = (const float*)d_in[3];
  const float* tg         = (const float*)d_in[4];
  const float* fg         = (const float*)d_in[5];
  const float* emb_s      = (const float*)d_in[6];
  const float* emb_r      = (const float*)d_in[7];
  const float* rad_w1     = (const float*)d_in[8];
  const float* rad_b1     = (const float*)d_in[9];
  const float* rad_ln1s   = (const float*)d_in[10];
  const float* rad_ln1b   = (const float*)d_in[11];
  const float* rad_w2     = (const float*)d_in[12];
  const float* rad_b2     = (const float*)d_in[13];
  const float* rad_ln2s   = (const float*)d_in[14];
  const float* rad_ln2b   = (const float*)d_in[15];
  const float* rad_w3     = (const float*)d_in[16];
  const float* rad_b3     = (const float*)d_in[17];
  const float* c1_w0      = (const float*)d_in[18];
  const float* c1_b0      = (const float*)d_in[19];
  const float* c1_w1      = (const float*)d_in[20];
  const float* c1_w2      = (const float*)d_in[21];
  const float* c2_w0      = (const float*)d_in[22];
  const float* c2_b0      = (const float*)d_in[23];
  const float* c2_w1      = (const float*)d_in[24];
  const float* c2_w2      = (const float*)d_in[25];
  const float* an_scale   = (const float*)d_in[26];
  const float* an_bias    = (const float*)d_in[27];
  const float* alpha_dot  = (const float*)d_in[28];
  const float* proj_w     = (const float*)d_in[29];
  const float* proj_b     = (const float*)d_in[30];
  const int*   species    = (const int*)d_in[31];
  const int*   senders    = (const int*)d_in[32];
  const int*   receivers  = (const int*)d_in[33];
  float* out = (float*)d_out;

  // workspace layout (floats). rad region is reused for node accumulators
  // after SO2-conv1 consumes it.
  float* ws = (float*)d_ws;
  float*    rad      = ws;                                    // E*1536
  float*    node_out = ws;                                    // N*25*128 (reuse)
  unsigned* nmax     = (unsigned*)(ws + 16000000);            // N*8
  float*    nden     = ws + 16000000 + 40000;                 // N*8
  float*    a_ex     = ws + 16000000 + 80000;                 // E*8
  float*    ef1      = ws + 38400000;                         // E*19*128 (also ef2)
  float*    efm      = ws + 38400000 + 60800000;              // E*19*64
  float*    x0a      = efm + 30400000;                        // E*256
  float*    x0g      = x0a + 6400000;                         // E*64
  float*    ef2      = ef1;

  // swizzled f16 weights (fragment-contiguous); offsets in halves
  _Float16* swz = (_Float16*)(ws + 137600000);
  _Float16* z_rw1  = swz;                  // 192*64   = 12288
  _Float16* z_rw2  = z_rw1 + 12288;        // 64*64    = 4096
  _Float16* z_rw3  = z_rw2 + 4096;         // 64*1536  = 98304
  _Float16* z_c1w0 = z_rw3 + 98304;        // 640*640  = 409600
  _Float16* z_c1w1 = z_c1w0 + 409600;      // 512*512  = 262144
  _Float16* z_c1w2 = z_c1w1 + 262144;      // 384*384  = 147456
  _Float16* z_c2w0 = z_c1w2 + 147456;      // 320*640  = 204800
  _Float16* z_c2w1 = z_c2w0 + 204800;      // 256*1024 = 262144
  _Float16* z_c2w2 = z_c2w1 + 262144;      // 192*768  = 147456
  _Float16* z_pw   = z_c2w2 + 147456;      // 5*128*64 = 40960

  // weight pre-swizzle pass
  k_swz<<<64, 256, 0, stream>>>(rad_w1, z_rw1, 192, 64);
  k_swz<<<64, 256, 0, stream>>>(rad_w2, z_rw2, 64, 64);
  k_swz<<<128, 256, 0, stream>>>(rad_w3, z_rw3, 64, 1536);
  k_swz<<<512, 256, 0, stream>>>(c1_w0, z_c1w0, 640, 640);
  k_swz<<<512, 256, 0, stream>>>(c1_w1, z_c1w1, 512, 512);
  k_swz<<<256, 256, 0, stream>>>(c1_w2, z_c1w2, 384, 384);
  k_swz<<<256, 256, 0, stream>>>(c2_w0, z_c2w0, 320, 640);
  k_swz<<<512, 256, 0, stream>>>(c2_w1, z_c2w1, 256, 1024);
  k_swz<<<256, 256, 0, stream>>>(c2_w2, z_c2w2, 192, 768);
  for (int l = 0; l < 5; ++l)
    k_swz<<<64, 256, 0, stream>>>(proj_w + (size_t)l * 8192, z_pw + (size_t)l * 8192, 128, 64);

  int etiles = (N_EDGES_ + 15) / 16;   // 1563
  int ntiles = (N_NODES_ + 15) / 16;   // 313

  k_radial<<<etiles, 32, 0, stream>>>(edist, emb_s, emb_r,
                                      z_rw1, rad_b1, rad_ln1s, rad_ln1b,
                                      z_rw2, rad_b2, rad_ln2s, rad_ln2b,
                                      z_rw3, rad_b3, species, senders, receivers, rad);
  k_wigner<<<N_EDGES_, 32, 0, stream>>>(node_feats, wig, senders, receivers, ef1);
  k_c1<<<etiles, 32, 0, stream>>>(ef1, rad, z_c1w0, c1_b0, z_c1w1, z_c1w2, efm, x0a, x0g);
  k_grid<<<N_EDGES_, 32, 0, stream>>>(tg, fg, x0g, efm);
  k_c2<<<etiles, 32, 0, stream>>>(efm, z_c2w0, c2_b0, z_c2w1, z_c2w2, ef2);
  k_init<<<4096, 256, 0, stream>>>(node_out, nmax, nden);
  k_alpha<<<(N_EDGES_ * NH_ + 255) / 256, 256, 0, stream>>>(x0a, an_scale, an_bias,
                                                            alpha_dot, receivers, a_ex, nmax);
  k_expsum<<<(N_EDGES_ * NH_ + 255) / 256, 256, 0, stream>>>(receivers, nmax, a_ex, nden);
  k_attn<<<N_EDGES_, 32, 0, stream>>>(ef2, winv, a_ex, nden, receivers, node_out);
  k_proj<<<ntiles, 32, 0, stream>>>(node_out, proj_w ? z_pw : z_pw, proj_b, out);

  (void)in_sizes; (void)n_in; (void)out_size; (void)ws_size;
}
